// canny_Net_53498112638992
// MI455X (gfx1250) — matrix operand order repaired and checked by reading, not compiled, not run on hardware
//
#include <hip/hip_runtime.h>
#include <cstdint>

// ---------------------------------------------------------------------------
// Fused Canny "early_threshold" for MI455X (gfx1250, wave32).
// out = where(|grad(gauss5x5 * img)| < 2, 0, |grad|)  -- the reference's NMS /
// orientation / direction-conv outputs are dead code and skipped entirely.
//
// Per 64x64 output tile (256 threads = 8 wave32):
//   0) zero hb/bl LDS regions (zero `SAME` padding + finite WMMA slack);
//      s_in zeroed only on border tiles (interior tiles overwrite it fully)
//   1) global_load_async_to_lds_b128 stage of 70x72 input tile (ASYNCcnt path)
//   2) horizontal 5-tap gaussian: banded 16x16 WMMA (A via aligned ds_load_b128)
//   3) vertical   5-tap gaussian: banded 16x16 WMMA (constant banded A-frag)
//   4) rolling-window Sobel + sqrt + threshold, coalesced stores
// Index spaces (out-relative):  s_in col c -> x=c-4, row r -> y=r-3
//                               s_hb col c -> x=c-2, row r -> y=r-3
//                               s_bl col c -> x=c-2, row r -> y=r-1
// ---------------------------------------------------------------------------

typedef _Float16 v16h __attribute__((ext_vector_type(16)));
typedef float    v8f  __attribute__((ext_vector_type(8)));

#define IMG_W   1024
#define IMG_H   1024
#define TILE    64
#define INS     72                 // staged width/stride: 18 float4 per row
#define INROWS  70
#define HBS     68
#define HBROWS  70
#define BLS     68
#define BLROWS  66
#define LDS_IN_F (INROWS * INS)                    // 5040 floats
#define LDS_HB_F (HBROWS * HBS)                    // 4760 floats
#define LDS_BL_F (BLROWS * BLS)                    // 4488 floats
#define LDS_TOTAL (LDS_IN_F + LDS_HB_F + LDS_BL_F) // 14288 floats
#define DUMP_IDX  LDS_TOTAL                        // predicated-store dump slot

// LDS byte offset of a pointer into __shared__ (generic -> AS3 addrspacecast).
__device__ __forceinline__ unsigned lds_off(const void* p) {
    return (unsigned)(unsigned long long)(__attribute__((address_space(3))) const char*)p;
}

// Unnormalized gaussian taps g[d] = exp(-0.5*(d-2)^2), d in [0,4], else 0.
__device__ __forceinline__ float gtap(int d) {
    if (d < 0 || d > 4) return 0.0f;
    int ad = d - 2; ad = ad < 0 ? -ad : ad;
    return ad == 0 ? 1.0f : (ad == 1 ? 0.60653065971f : 0.13533528324f);
}

__global__ __launch_bounds__(256, 1)
void canny_early_thresh_kernel(const float* __restrict__ img,
                               float* __restrict__ out)
{
    __shared__ __align__(16) float lds[LDS_TOTAL + 4];
    float* s_in = lds;
    float* s_hb = lds + LDS_IN_F;
    float* s_bl = s_hb + LDS_HB_F;

    const int t    = threadIdx.x;
    const int lane = t & 31;
    // readfirstlane => compiler proves wave-uniformity => scalar chunk control flow
    const int wv   = __builtin_amdgcn_readfirstlane(t >> 5);   // wave 0..7
    const int hi   = lane >> 4;       // wave half (wave32 WMMA striping)
    const int nl   = lane & 15;       // M (A) / N (B,D) index
    const int tx0  = blockIdx.x * TILE;
    const int ty0  = blockIdx.y * TILE;
    const long long base = (long long)blockIdx.z * (long long)(IMG_W * IMG_H);
    const bool border = (tx0 == 0) | (tx0 == IMG_W - TILE) |
                        (ty0 == 0) | (ty0 == IMG_H - TILE);

    // ---- 0) zero hb/bl (b128). s_in only on border tiles (else fully staged).
    {
        float4 z4 = make_float4(0.f, 0.f, 0.f, 0.f);
        float4* zp = (float4*)s_hb;
        for (int i = t; i < (LDS_HB_F + LDS_BL_F) / 4; i += 256) zp[i] = z4;
        if (border) {
            float4* ip = (float4*)s_in;
            for (int i = t; i < LDS_IN_F / 4; i += 256) ip[i] = z4;
        }
    }
    __syncthreads();

    // ---- 1) async b128 stage: 70 rows x 18 float4 (halo: left 4, top 3).
    //         tx0 and IMG_W are multiples of 4 => every float4 is fully
    //         in-image or fully out; skipped vectors stay zero.
    for (int i = t; i < INROWS * 18; i += 256) {
        int r = i / 18, j = i - r * 18;
        int gy  = ty0 + r - 3;
        int gx4 = tx0 - 4 + 4 * j;
        if ((unsigned)gy < (unsigned)IMG_H && (unsigned)gx4 <= (unsigned)(IMG_W - 4)) {
            unsigned long long ga = (unsigned long long)(uintptr_t)
                (img + base + (long long)gy * IMG_W + gx4);
            unsigned la = lds_off(s_in + r * INS + 4 * j);
            asm volatile("global_load_async_to_lds_b128 %0, %1, off"
                         :: "v"(la), "v"(ga) : "memory");
        }
    }
    asm volatile("s_wait_asynccnt 0" ::: "memory");
    __syncthreads();

    // ---- WMMA fragments (f16 16x16x32, wave32; A and B share one k order per pass):
    //   horizontal A,B: lane<16 holds K {0..7,16..23}; lane>=16 holds K {8..15,24..31}
    //   vertical A,B (contiguous): lane<16 holds K 0..15; lane>=16 holds K 16..31; M/N = lane&15
    //   D: VGPR v -> M = v + 8*hi, N = lane&15
    v16h bfragH;   // horizontal pass B: B(k,n) = g[k-n]
    v16h afragV;   // vertical   pass A: A(m,k) = g[k-m]
    #pragma unroll
    for (int e = 0; e < 16; ++e) {
        int kB = (e < 8) ? (hi * 8 + e) : (16 + hi * 8 + (e - 8));
        bfragH[e] = (_Float16)gtap(kB - nl);
        int kA = hi * 16 + e;
        afragV[e] = (_Float16)gtap(kA - nl);
    }

    // Block-uniform validity windows (collapse rr-limit + gy/gx image guard
    // into one unsigned range compare):
    const int rlo1 = (ty0 == 0) ? 3 : 0;                    // hb rows: gy >= 0
    const int rhi1 = (ty0 == IMG_H - TILE) ? 67 : HBROWS;   // gy < 1024, rr < 70
    const int rlo2 = (ty0 == 0) ? 1 : 0;                    // bl rows
    const int rhi2 = (ty0 == IMG_H - TILE) ? 65 : BLROWS;
    const int clo2 = (tx0 == 0) ? 2 : 0;                    // bl cols (gx guard)
    const int chi2 = (tx0 == IMG_W - TILE) ? 66 : BLS;

    // ---- 2) horizontal blur: hb[r][c] = sum_i g[i]*in[r][c+i]
    for (int q = wv; q < 25; q += 8) {
        int r0 = (q / 5) * 16, c0 = (q % 5) * 16;           // scalar
        const float4* ap = (const float4*)(s_in + (r0 + nl) * INS + c0);
        float4 q0 = ap[hi * 2];        // cols c0 + hi*8 + 0..3
        float4 q1 = ap[hi * 2 + 1];    // cols c0 + hi*8 + 4..7
        float4 q2 = ap[hi * 2 + 4];    // cols c0 + 16 + hi*8 + 0..3
        float4 q3 = ap[hi * 2 + 5];    // cols c0 + 16 + hi*8 + 4..7
        v16h a;
        a[0]  = (_Float16)q0.x; a[1]  = (_Float16)q0.y; a[2]  = (_Float16)q0.z; a[3]  = (_Float16)q0.w;
        a[4]  = (_Float16)q1.x; a[5]  = (_Float16)q1.y; a[6]  = (_Float16)q1.z; a[7]  = (_Float16)q1.w;
        a[8]  = (_Float16)q2.x; a[9]  = (_Float16)q2.y; a[10] = (_Float16)q2.z; a[11] = (_Float16)q2.w;
        a[12] = (_Float16)q3.x; a[13] = (_Float16)q3.y; a[14] = (_Float16)q3.z; a[15] = (_Float16)q3.w;
        v8f acc = {};
        acc = __builtin_amdgcn_wmma_f32_16x16x32_f16(false, a, false, bfragH,
                                                     (short)0, acc, false, false);
        int rbase = r0 + hi * 8;
        int didx  = LDS_IN_F + rbase * HBS + c0 + nl;
        if (r0 <= 48 && c0 <= 48 && r0 >= rlo1) {           // scalar branch
            #pragma unroll
            for (int v = 0; v < 8; ++v) lds[didx + v * HBS] = acc[v];
        } else {
            // predicated dump-slot stores: no exec juggling, no branches.
            // (x-OOI hb entries only feed bl entries S2 guards out -> no gx test)
            bool colOK = (c0 + nl) < HBS;
            int  rb    = rbase - rlo1;
            #pragma unroll
            for (int v = 0; v < 8; ++v) {
                bool p = colOK & ((unsigned)(rb + v) < (unsigned)(rhi1 - rlo1));
                lds[p ? (didx + v * HBS) : DUMP_IDX] = acc[v];
            }
        }
    }
    __syncthreads();

    // ---- 3) vertical blur: bl[r][c] = sum_j g[j]*hb[r+j][c]
    for (int q = wv; q < 25; q += 8) {
        int r0 = (q / 5) * 16, c0 = (q % 5) * 16;           // scalar
        int bidx = (r0 + hi * 16) * HBS + c0 + nl;
        v16h b;
        #pragma unroll
        for (int e = 0; e < 16; ++e)           // 16 ds_load_b32, immediate offsets
            b[e] = (_Float16)s_hb[bidx + e * HBS];
        v8f acc = {};
        acc = __builtin_amdgcn_wmma_f32_16x16x32_f16(false, afragV, false, b,
                                                     (short)0, acc, false, false);
        int rbase = r0 + hi * 8;
        int didx  = LDS_IN_F + LDS_HB_F + rbase * BLS + c0 + nl;
        if (r0 <= 48 && c0 <= 48 && r0 >= rlo2 && c0 >= clo2) {   // scalar branch
            #pragma unroll
            for (int v = 0; v < 8; ++v) lds[didx + v * BLS] = acc[v];
        } else {
            int  cc    = c0 + nl;
            bool colOK = (unsigned)(cc - clo2) < (unsigned)(chi2 - clo2);
            int  rb    = rbase - rlo2;
            #pragma unroll
            for (int v = 0; v < 8; ++v) {
                bool p = colOK & ((unsigned)(rb + v) < (unsigned)(rhi2 - rlo2));
                lds[p ? (didx + v * BLS) : DUMP_IDX] = acc[v];
            }
        }
    }
    __syncthreads();

    // ---- 4) Sobel + magnitude + threshold; rolling window over 16 rows.
    //         Output (yy,x) taps bl rows {yy,yy+1,yy+2}, cols {x+1,x+2,x+3}.
    {
        const int x  = t & 63;
        const int y0 = (t >> 6) * 16;
        const float* bp = s_bl + y0 * BLS + (x + 1);
        float a0 = bp[0], a1 = bp[1], a2 = bp[2];
        float s0 = a0 + 2.0f * a1 + a2, d0 = a0 - a2;
        a0 = bp[BLS]; a1 = bp[BLS + 1]; a2 = bp[BLS + 2];
        float s1 = a0 + 2.0f * a1 + a2, d1 = a0 - a2;
        float* orow = out + base + (long long)(ty0 + y0) * IMG_W + (tx0 + x);
        #pragma unroll
        for (int k = 0; k < 16; ++k) {
            const float* rp = bp + (k + 2) * BLS;
            a0 = rp[0]; a1 = rp[1]; a2 = rp[2];
            float s2 = a0 + 2.0f * a1 + a2, d2 = a0 - a2;
            float gxv = d0 + 2.0f * d1 + d2;     // col(x-1) - col(x+1), rows [1,2,1]
            float gyv = s0 - s2;                 // row(y-1) - row(y+1), cols [1,2,1]
            float mag = sqrtf(gxv * gxv + gyv * gyv);
            orow[(long long)k * IMG_W] = (mag < 2.0f) ? 0.0f : mag;
            s0 = s1; s1 = s2; d0 = d1; d1 = d2;
        }
    }
}

extern "C" void kernel_launch(void* const* d_in, const int* in_sizes, int n_in,
                              void* d_out, int out_size, void* d_ws, size_t ws_size,
                              hipStream_t stream) {
    const float* img = (const float*)d_in[0];
    float*       o   = (float*)d_out;
    (void)d_ws; (void)ws_size; (void)n_in; (void)out_size;

    int N = in_sizes[0] / (IMG_W * IMG_H);     // batch (8 in reference)
    dim3 grid(IMG_W / TILE, IMG_H / TILE, N);  // 16 x 16 x N tiles
    dim3 block(256);                           // 8 wave32s
    hipLaunchKernelGGL(canny_early_thresh_kernel, grid, block, 0, stream, img, o);
}
